// DynamicBottleneck_45311904973379
// MI455X (gfx1250) — matrix operand order repaired and checked by reading, not compiled, not run on hardware
//
#include <hip/hip_runtime.h>
#include <hip/hip_bf16.h>

// ---------------- problem constants ----------------
#define BATCH   8
#define CH      256
#define HH      80
#define WW      80
#define HW      (HH*WW)            // 6400
#define NPIX    (BATCH*HW)         // 51200
#define KDIM    (CH*9)             // 2304
#define GROUPS  32
#define CPG     (CH/GROUPS)        // 8
#define GN_EPS  1e-5f

// GEMM tiling
#define BM 128      // co per block
#define BN 128      // pixels per block
#define BK 32       // K per step
#define KSTEPS (KDIM/BK)   // 72

typedef __attribute__((ext_vector_type(16))) __bf16 v16bf;
typedef __attribute__((ext_vector_type(8)))  __bf16 v8bf;
typedef __attribute__((ext_vector_type(8)))  float  v8f;

#if __has_builtin(__builtin_amdgcn_global_load_async_to_lds_b128) && \
    __has_builtin(__builtin_amdgcn_s_wait_asynccnt)
#define USE_ASYNC 1
typedef int ivec4 __attribute__((vector_size(16)));
typedef __attribute__((address_space(1))) ivec4 g_iv4;   // global int4
typedef __attribute__((address_space(3))) ivec4 l_iv4;   // LDS int4
#else
#define USE_ASYNC 0
#endif

// ---- bf16 helpers via bit ops ----
static __device__ __forceinline__ __bf16 f2bf(float f) {
    union { float f; unsigned u; } v; v.f = f;
    unsigned r = v.u + 0x7FFFu + ((v.u >> 16) & 1u);   // round-to-nearest-even
    unsigned short h = (unsigned short)(r >> 16);
    __bf16 o; __builtin_memcpy(&o, &h, 2); return o;
}
static __device__ __forceinline__ float bf2f(__bf16 b) {
    unsigned short h; __builtin_memcpy(&h, &b, 2);
    union { unsigned u; float f; } v; v.u = ((unsigned)h) << 16; return v.f;
}

// ---------------- 0) zero scratch (async-copy padding source) ----------------
__global__ void k_zero(float* __restrict__ z) { z[threadIdx.x] = 0.0f; }

// ---------------- 1) pack x (NCHW f32) -> xh (NHWC bf16) ----------------
__global__ void k_pack_x_nhwc(const float* __restrict__ x, __bf16* __restrict__ xh) {
    int idx = blockIdx.x * blockDim.x + threadIdx.x;       // NHWC linear
    int c   = idx & (CH - 1);
    int rest = idx >> 8;
    int hw  = rest % HW;
    int b   = rest / HW;
    xh[idx] = f2bf(x[(b * CH + c) * HW + hw]);
}

// ---------------- 2) pack w (O,I,3,3 f32) -> wt bf16 [co][ (r*3+s)*C + ci ] ----
__global__ void k_pack_w(const float* __restrict__ w, __bf16* __restrict__ wt) {
    int idx = blockIdx.x * blockDim.x + threadIdx.x;       // co*KDIM + k
    int k  = idx % KDIM;
    int co = idx / KDIM;
    int rs = k >> 8;                 // 0..8
    int ci = k & (CH - 1);
    wt[idx] = f2bf(w[(co * CH + ci) * 9 + rs]);
}

// ---------------- 3) spatial gate: wave per pixel ----------------
__global__ void k_gate(const __bf16* __restrict__ xh, const float* __restrict__ gw,
                       const float* __restrict__ gb, float* __restrict__ gate) {
    int tid  = threadIdx.x;
    int lane = tid & 31;
    int pix  = blockIdx.x * 8 + (tid >> 5);
    const v8bf xv = *reinterpret_cast<const v8bf*>(xh + (size_t)pix * CH + lane * 8);
    float s = 0.f;
#pragma unroll
    for (int e = 0; e < 8; ++e)
        s += bf2f(xv[e]) * gw[lane * 8 + e];
#pragma unroll
    for (int off = 16; off > 0; off >>= 1)
        s += __shfl_xor(s, off, 32);
    if (lane == 0) {
        float g = s + gb[0];
        gate[pix] = fmaxf(tanhf(g), 0.0f);
    }
}

// ---------------- 4) implicit-GEMM conv via WMMA bf16 ----------------
// xh: NHWC bf16, wt: [256][2304] bf16, zbuf: 256B of zeros, out: NCHW f32
__global__ __launch_bounds__(256, 2)
void k_conv_wmma(const __bf16* __restrict__ xh, const __bf16* __restrict__ wt,
                 const __bf16* __restrict__ zbuf, float* __restrict__ out) {
    __shared__ __bf16 As[2][BM][BK];    // 16 KB: [buf][co][k]
    __shared__ __bf16 Bs[2][BN][BK];    // 16 KB: [buf][pixel][k]

    const int tid    = threadIdx.x;
    const int lane   = tid & 31;
    const int half   = lane >> 4;          // 0 or 1
    const int lm     = lane & 15;
    const int waveid = tid >> 5;           // 0..7
    const int m0     = (waveid & 3) * 32;  // 4 wave-rows of 32 co
    const int n0     = (waveid >> 2) * 64; // 2 wave-cols of 64 px

    const int coBase = blockIdx.y * BM;
    const int pBase  = blockIdx.x * BN;
    const int bIdx   = pBase / HW;         // tile never crosses batch (6400 % 128 == 0)
    const int hwBase = pBase % HW;

    // per-thread staging assignment: both tiles are 128 rows x 32 k (64B rows)
    const int row  = tid >> 1;             // 0..127
    const int chk  = (tid & 1) * 16;       // element offset (16 bf16 = 32B)
    const int hB   = (hwBase + row) / WW;
    const int wB   = (hwBase + row) % WW;
    const __bf16* aSrcBase = wt + (size_t)(coBase + row) * KDIM + chk;

    // stage K-tile kt into LDS buffer `buf`
    auto stage = [&](int kt, int buf) {
        const int k0  = kt * BK;
        const int rs  = k0 >> 8;
        const int ci0 = k0 & (CH - 1);
        const int r   = rs / 3, s = rs % 3;
        const int ih = hB + r - 1, iw = wB + s - 1;
        const bool ok = (ih >= 0 && ih < HH && iw >= 0 && iw < WW);
        const __bf16* aSrc = aSrcBase + k0;
        const __bf16* bSrc = ok
            ? xh + ((size_t)(bIdx * HW + ih * WW + iw) * CH + ci0) + chk
            : zbuf;                          // zeros for conv padding
#if USE_ASYNC
        __builtin_amdgcn_global_load_async_to_lds_b128(
            (g_iv4*)aSrc, (l_iv4*)&As[buf][row][chk], 0, 0);
        __builtin_amdgcn_global_load_async_to_lds_b128(
            (g_iv4*)aSrc, (l_iv4*)&As[buf][row][chk], 16, 0);
        __builtin_amdgcn_global_load_async_to_lds_b128(
            (g_iv4*)bSrc, (l_iv4*)&Bs[buf][row][chk], 0, 0);
        __builtin_amdgcn_global_load_async_to_lds_b128(
            (g_iv4*)bSrc, (l_iv4*)&Bs[buf][row][chk], 16, 0);
#else
        {
            const uint4* s4 = reinterpret_cast<const uint4*>(aSrc);
            uint4 v0 = s4[0], v1 = s4[1];
            uint4* d4 = reinterpret_cast<uint4*>(&As[buf][row][chk]);
            d4[0] = v0; d4[1] = v1;
        }
        {
            const uint4* s4 = reinterpret_cast<const uint4*>(bSrc);
            uint4 v0 = s4[0], v1 = s4[1];
            uint4* d4 = reinterpret_cast<uint4*>(&Bs[buf][row][chk]);
            d4[0] = v0; d4[1] = v1;
        }
#endif
    };

    v8f acc[2][4] = {};

    stage(0, 0);
    stage(1, 1);

    for (int kt = 0; kt < KSTEPS; ++kt) {
        const int buf = kt & 1;
#if USE_ASYNC
        if (kt + 1 < KSTEPS) __builtin_amdgcn_s_wait_asynccnt(4);  // oldest tile done
        else                 __builtin_amdgcn_s_wait_asynccnt(0);
#endif
        __syncthreads();     // all waves' tile-kt data visible in LDS

        // fragments: A per ISA 16-bit 16x32 layout, B per 32x16 layout
        v16bf afrag[2], bfrag[4];
#pragma unroll
        for (int i = 0; i < 2; ++i) {
            const int rr = m0 + 16 * i + lm;
            v8bf lo = *reinterpret_cast<const v8bf*>(&As[buf][rr][half * 8]);
            v8bf hi = *reinterpret_cast<const v8bf*>(&As[buf][rr][16 + half * 8]);
#pragma unroll
            for (int e = 0; e < 8; ++e) { afrag[i][e] = lo[e]; afrag[i][e + 8] = hi[e]; }
        }
#pragma unroll
        for (int j = 0; j < 4; ++j) {
            const int rn = n0 + 16 * j + lm;
            v8bf lo = *reinterpret_cast<const v8bf*>(&Bs[buf][rn][half * 8]);
            v8bf hi = *reinterpret_cast<const v8bf*>(&Bs[buf][rn][16 + half * 8]);
#pragma unroll
            for (int e = 0; e < 8; ++e) { bfrag[j][e] = lo[e]; bfrag[j][e + 8] = hi[e]; }
        }
#pragma unroll
        for (int i = 0; i < 2; ++i)
#pragma unroll
            for (int j = 0; j < 4; ++j)
                acc[i][j] = __builtin_amdgcn_wmma_f32_16x16x32_bf16(
                    false, afrag[i], false, bfrag[j], (short)0, acc[i][j], false, false);

        __syncthreads();     // everyone done reading buf before refill
        if (kt + 2 < KSTEPS) stage(kt + 2, buf);
    }

    // --- store D: lane L, vgpr r -> (M = r + 8*half, N = lm) ---
#pragma unroll
    for (int i = 0; i < 2; ++i)
#pragma unroll
        for (int j = 0; j < 4; ++j) {
            const int hw = hwBase + n0 + 16 * j + lm;
#pragma unroll
            for (int rr = 0; rr < 8; ++rr) {
                const int co = coBase + m0 + 16 * i + rr + 8 * half;
                out[((size_t)(bIdx * CH + co)) * HW + hw] = acc[i][j][rr];
            }
        }
}

// ---------------- 5) GroupNorm stats: one block per (b, group) ----------------
__global__ void k_gn_stats(const float* __restrict__ h, float* __restrict__ stats) {
    __shared__ float s_sum[256];
    __shared__ float s_sq[256];
    const int b = blockIdx.x / GROUPS;
    const int g = blockIdx.x % GROUPS;
    const int tid = threadIdx.x;
    const size_t base = ((size_t)b * CH + g * CPG) * HW;
    float sum = 0.f, sq = 0.f;
    for (int i = tid; i < CPG * HW; i += 256) {
        float v = h[base + i];
        sum += v; sq += v * v;
    }
    s_sum[tid] = sum; s_sq[tid] = sq;
    __syncthreads();
    for (int off = 128; off > 0; off >>= 1) {
        if (tid < off) { s_sum[tid] += s_sum[tid + off]; s_sq[tid] += s_sq[tid + off]; }
        __syncthreads();
    }
    if (tid == 0) {
        const float n = (float)(CPG * HW);
        float mean = s_sum[0] / n;
        float var  = s_sq[0] / n - mean * mean;
        stats[blockIdx.x * 2 + 0] = mean;
        stats[blockIdx.x * 2 + 1] = rsqrtf(var + GN_EPS);
    }
}

// ---------------- 6) GN apply + ReLU + repack to NHWC bf16 ----------------
__global__ void k_gn_relu_pack(const float* __restrict__ h, const float* __restrict__ stats,
                               const float* __restrict__ gnw, const float* __restrict__ gnb,
                               __bf16* __restrict__ xh2) {
    int idx = blockIdx.x * blockDim.x + threadIdx.x;   // NHWC linear
    int c    = idx & (CH - 1);
    int rest = idx >> 8;
    int hw   = rest % HW;
    int b    = rest / HW;
    int sg   = b * GROUPS + (c >> 3);
    float mean = stats[sg * 2 + 0], inv = stats[sg * 2 + 1];
    float v = h[((size_t)(b * CH + c)) * HW + hw];
    v = (v - mean) * inv * gnw[c] + gnb[c];
    xh2[idx] = f2bf(fmaxf(v, 0.0f));
}

// ---------------- 7) final: GN2 * gate + shortcut, ReLU ----------------
__global__ void k_final(const float* __restrict__ h2, const float* __restrict__ stats,
                        const float* __restrict__ gnw, const float* __restrict__ gnb,
                        const float* __restrict__ gate, const float* __restrict__ x,
                        float* __restrict__ out) {
    size_t idx = (size_t)blockIdx.x * blockDim.x + threadIdx.x;  // NCHW linear
    int hw = (int)(idx % HW);
    int c  = (int)((idx / HW) % CH);
    int b  = (int)(idx / ((size_t)HW * CH));
    int sg = b * GROUPS + (c >> 3);
    float mean = stats[sg * 2 + 0], inv = stats[sg * 2 + 1];
    float v = (h2[idx] - mean) * inv * gnw[c] + gnb[c];
    v = v * gate[b * HW + hw] + x[idx];
    out[idx] = fmaxf(v, 0.0f);
}

// ---------------- launch ----------------
extern "C" void kernel_launch(void* const* d_in, const int* in_sizes, int n_in,
                              void* d_out, int out_size, void* d_ws, size_t ws_size,
                              hipStream_t stream) {
    (void)in_sizes; (void)n_in; (void)out_size; (void)ws_size;
    const float* x     = (const float*)d_in[0];
    const float* w1    = (const float*)d_in[1];
    const float* gn1w  = (const float*)d_in[2];
    const float* gn1b  = (const float*)d_in[3];
    const float* w2    = (const float*)d_in[4];
    const float* gn2w  = (const float*)d_in[5];
    const float* gn2b  = (const float*)d_in[6];
    const float* gw    = (const float*)d_in[7];
    const float* gb    = (const float*)d_in[8];
    float* out = (float*)d_out;

    char* ws = (char*)d_ws;
    __bf16* xh    = (__bf16*)(ws + 0);                 // 26,214,400 B
    __bf16* wt1   = (__bf16*)(ws + 26214400);          //  1,179,648 B
    __bf16* wt2   = (__bf16*)(ws + 27394048);          //  1,179,648 B
    float*  h1    = (float*)(ws + 28573696);           // 52,428,800 B
    __bf16* xh2   = (__bf16*)(ws + 81002496);          // 26,214,400 B
    float*  h2    = (float*)(ws + 107216896);          // 52,428,800 B
    float*  gate  = (float*)(ws + 159645696);          //    204,800 B
    float*  st1   = (float*)(ws + 159850496);          //      2,048 B
    float*  st2   = (float*)(ws + 159852544);          //      2,048 B
    float*  zbuf  = (float*)(ws + 159854592);          //        256 B

    const int NEL = BATCH * CH * HW;                   // 13,107,200

    k_zero<<<1, 64, 0, stream>>>(zbuf);
    k_pack_x_nhwc<<<NEL / 256, 256, 0, stream>>>(x, xh);
    k_pack_w<<<(CH * KDIM) / 256, 256, 0, stream>>>(w1, wt1);
    k_pack_w<<<(CH * KDIM) / 256, 256, 0, stream>>>(w2, wt2);
    k_gate<<<NPIX / 8, 256, 0, stream>>>(xh, gw, gb, gate);

    dim3 cgrid(NPIX / BN, CH / BM);                    // (400, 2)
    k_conv_wmma<<<cgrid, 256, 0, stream>>>(xh, wt1, (const __bf16*)zbuf, h1);
    k_gn_stats<<<BATCH * GROUPS, 256, 0, stream>>>(h1, st1);
    k_gn_relu_pack<<<NEL / 256, 256, 0, stream>>>(h1, st1, gn1w, gn1b, xh2);

    k_conv_wmma<<<cgrid, 256, 0, stream>>>(xh2, wt2, (const __bf16*)zbuf, h2);
    k_gn_stats<<<BATCH * GROUPS, 256, 0, stream>>>(h2, st2);
    k_final<<<NEL / 256, 256, 0, stream>>>(h2, st2, gn2w, gn2b, gate, x, out);
}
